// MPNNLayer_32272384262602
// MI455X (gfx1250) — matrix operand order repaired and checked by reading, not compiled, not run on hardware
//
#include <hip/hip_runtime.h>
#include <hip/hip_bf16.h>
#include <math.h>

// MPNN layer for MI455X (gfx1250, wave32, WMMA bf16 16x16x32).
// edge MLP (3 GEMM layers, erf-GELU) + atomic scatter-sum + node MLP + 2x LayerNorm.
// Weights are pre-swizzled into per-lane WMMA B-fragment order so each fragment
// is two global_load_b128 from L2 instead of 16 scalar u16 loads.

typedef __attribute__((ext_vector_type(16))) __bf16 v16bf;
typedef __attribute__((ext_vector_type(8)))  float  v8f;
typedef __attribute__((ext_vector_type(4)))  float  f4v;
typedef __bf16 bf16;

#define N_NODES 20000
#define N_EDGES 320000
#define INV_SCALE (1.0f / 30.0f)

union FragU { v16bf v; unsigned u32[8]; };
union FragQ { v16bf v; uint4 q[2]; };

__device__ inline unsigned pack2(float a, float b) {
  union { __bf16 h[2]; unsigned u; } t;
  t.h[0] = (__bf16)a; t.h[1] = (__bf16)b;
  return t.u;
}

__device__ inline float gelu_f(float x) {
  return 0.5f * x * (1.0f + erff(x * 0.70710678118654752f));
}

__device__ inline v8f wmma_bf16(v16bf a, v16bf b, v8f c) {
  return __builtin_amdgcn_wmma_f32_16x16x32_bf16(false, a, false, b, (short)0, c,
                                                 false, false);
}

// A fragment (16x32 bf16, M x K) from LDS, row-major, leading dim ld.
// lane 0-15: row M = l, K in {k0..k0+7, k0+16..k0+23}; lanes 16-31: +8 on K groups.
// Lowers to 2x ds_load_b128 per fragment.
__device__ inline v16bf frag_a_lds(const __bf16* buf, int ld, int row0, int k0, int lane) {
  const int l = lane & 15, hi = (lane >> 4) & 1;
  const __bf16* p = buf + (row0 + l) * ld + k0 + hi * 8;
  FragU f;
  f.u32[0] = *(const unsigned*)(p + 0);
  f.u32[1] = *(const unsigned*)(p + 2);
  f.u32[2] = *(const unsigned*)(p + 4);
  f.u32[3] = *(const unsigned*)(p + 6);
  f.u32[4] = *(const unsigned*)(p + 16);
  f.u32[5] = *(const unsigned*)(p + 18);
  f.u32[6] = *(const unsigned*)(p + 20);
  f.u32[7] = *(const unsigned*)(p + 22);
  return f.v;
}

// B fragment (32x16, K x N) from a pre-swizzled weight buffer: tiles of 32x16
// stored tile-major ((k0/32)*ntiles + n0/16), 512 bf16 per tile, within a tile
// lane-major with each lane's 16 bf16 contiguous. -> 2x global_load_b128.
__device__ inline v16bf frag_b_swz(const bf16* __restrict__ Bsw, int ntiles, int n0,
                                   int k0, int lane) {
  const int tile = (k0 >> 5) * ntiles + (n0 >> 4);
  const uint4* p = (const uint4*)(Bsw + (size_t)tile * 512 + lane * 16);
  FragQ f;
  f.q[0] = p[0];
  f.q[1] = p[1];
  return f.v;
}

// ---------------------------------------------------------------------------
// prep kernels
// ---------------------------------------------------------------------------
__global__ __launch_bounds__(256) void zero_kernel(float4* p, int n4) {
  int i = blockIdx.x * 256 + threadIdx.x;
  if (i < n4) p[i] = float4{0.f, 0.f, 0.f, 0.f};
}

// Convert fp32 (K x N, row-major) weight to bf16 in WMMA B-fragment swizzled
// order. Fragment layout: lane = hi*16 + l; lane's j-th element = W[k][n] with
// k = ktile*32 + hi*8 + (j&7) + (j>>3)*16, n = ntile*16 + l.
__global__ __launch_bounds__(256) void swz_kernel(const float* __restrict__ s,
                                                  bf16* __restrict__ d, int K, int N) {
  int o = blockIdx.x * 256 + threadIdx.x;
  if (o >= K * N) return;
  int tile = o >> 9;       // 512 elements per 32x16 tile
  int rem  = o & 511;
  int lane = rem >> 4;
  int j    = rem & 15;
  int ntiles = N >> 4;
  int kt = tile / ntiles;
  int nt = tile - kt * ntiles;
  int hi = lane >> 4, l = lane & 15;
  int k = kt * 32 + hi * 8 + (j & 7) + ((j >> 3) << 4);
  int n = nt * 16 + l;
  d[o] = (__bf16)s[k * N + n];
}

// ---------------------------------------------------------------------------
// edge kernel: h_msg = W3(gelu(W2(gelu(W1(h_E))))); dh[src] += h_msg/30
// 2500 blocks x 256 threads, 128 edges per block. LDS: 64KB + 32KB.
// ---------------------------------------------------------------------------
__global__ __launch_bounds__(256) void edge_mlp_scatter(
    const float* __restrict__ hE, const int* __restrict__ src_idx,
    const bf16* __restrict__ W1b, const float* __restrict__ b1,
    const bf16* __restrict__ W2b, const float* __restrict__ b2,
    const bf16* __restrict__ W3b, const float* __restrict__ b3,
    float* __restrict__ dh) {
  __shared__ __bf16 ebuf[128 * 256];  // input tile (layer1 A); reused as layer2 out
  __shared__ __bf16 abuf[128 * 128];  // layer1 out / layer2 A

  const int tid = threadIdx.x;
  const int lane = tid & 31, wave = tid >> 5;
  const int l = lane & 15, hi = (lane >> 4) & 1;
  const long e0 = (long)blockIdx.x * 128;
  const int m0 = wave * 16;

  // ---- stage h_E tile, fp32 -> bf16, coalesced non-temporal float4 loads ----
  for (int i = tid; i < 128 * 64; i += 256) {
    int row = i >> 6, c4 = (i & 63) << 2;
    f4v f = __builtin_nontemporal_load((const f4v*)(hE + (e0 + row) * 256 + c4));
    unsigned* dst = (unsigned*)(&ebuf[row * 256 + c4]);
    dst[0] = pack2(f.x, f.y);
    dst[1] = pack2(f.z, f.w);
  }
  __syncthreads();

  // ---- layer 1: (128x256) @ (256x128), K-outer for A reuse ----
  {
    v8f acc[8];
    v8f z = {};
#pragma unroll
    for (int nc = 0; nc < 8; ++nc) acc[nc] = z;
#pragma unroll
    for (int ks = 0; ks < 8; ++ks) {
      v16bf a = frag_a_lds(ebuf, 256, m0, ks * 32, lane);
#pragma unroll
      for (int nc = 0; nc < 8; ++nc) {
        v16bf b = frag_b_swz(W1b, 8, nc * 16, ks * 32, lane);
        acc[nc] = wmma_bf16(a, b, acc[nc]);
      }
    }
#pragma unroll
    for (int nc = 0; nc < 8; ++nc) {
      int col = nc * 16 + l;
      float bias = b1[col];
#pragma unroll
      for (int v = 0; v < 8; ++v) {
        int r = m0 + v + 8 * hi;
        abuf[r * 128 + col] = (__bf16)gelu_f(acc[nc][v] + bias);
      }
    }
  }
  __syncthreads();  // all waves past ebuf reads before layer-2 epilogue reuses it

  // ---- layer 2: (128x128) @ (128x128) -> gbuf (= ebuf reused, ld 128) ----
  __bf16* gbuf = ebuf;
  {
    v8f acc[8];
    v8f z = {};
#pragma unroll
    for (int nc = 0; nc < 8; ++nc) acc[nc] = z;
#pragma unroll
    for (int ks = 0; ks < 4; ++ks) {
      v16bf a = frag_a_lds(abuf, 128, m0, ks * 32, lane);
#pragma unroll
      for (int nc = 0; nc < 8; ++nc) {
        v16bf b = frag_b_swz(W2b, 8, nc * 16, ks * 32, lane);
        acc[nc] = wmma_bf16(a, b, acc[nc]);
      }
    }
#pragma unroll
    for (int nc = 0; nc < 8; ++nc) {
      int col = nc * 16 + l;
      float bias = b2[col];
#pragma unroll
      for (int v = 0; v < 8; ++v) {
        int r = m0 + v + 8 * hi;
        gbuf[r * 128 + col] = (__bf16)gelu_f(acc[nc][v] + bias);
      }
    }
  }
  __syncthreads();

  // ---- layer 3 + scaled atomic scatter into dh ----
  {
    v8f acc[8];
    v8f z = {};
#pragma unroll
    for (int nc = 0; nc < 8; ++nc) acc[nc] = z;
#pragma unroll
    for (int ks = 0; ks < 4; ++ks) {
      v16bf a = frag_a_lds(gbuf, 128, m0, ks * 32, lane);
#pragma unroll
      for (int nc = 0; nc < 8; ++nc) {
        v16bf b = frag_b_swz(W3b, 8, nc * 16, ks * 32, lane);
        acc[nc] = wmma_bf16(a, b, acc[nc]);
      }
    }
    int nid[8];
#pragma unroll
    for (int v = 0; v < 8; ++v) nid[v] = src_idx[e0 + m0 + v + 8 * hi];
#pragma unroll
    for (int nc = 0; nc < 8; ++nc) {
      int col = nc * 16 + l;
      float bias = b3[col];
#pragma unroll
      for (int v = 0; v < 8; ++v) {
        float val = (acc[nc][v] + bias) * INV_SCALE;
        atomicAdd(dh + (size_t)nid[v] * 128 + col, val);
      }
    }
  }
}

// ---------------------------------------------------------------------------
// node kernel: x = LN1(h_V + dh); y = gelu(x@d1+b)@d2+b; out = LN2(x + y)
// 157 blocks x 256 threads, 128 nodes per block. LDS: 32+32+64 KB.
// ---------------------------------------------------------------------------
__global__ __launch_bounds__(256) void node_update(
    const float* __restrict__ hV, const float* __restrict__ dh,
    const bf16* __restrict__ D1b, const float* __restrict__ db1,
    const bf16* __restrict__ D2b, const float* __restrict__ db2,
    const float* __restrict__ g1, const float* __restrict__ be1,
    const float* __restrict__ g2, const float* __restrict__ be2,
    float* __restrict__ out) {
  __shared__ __bf16 xa[128 * 128];    // LN1 output (bf16): GEMM1 A + residual
  __shared__ __bf16 ybuf[128 * 128];  // gelu intermediate ping (per 128-col group)
  __shared__ float rbuf[128 * 128];   // fp32 staging for both LayerNorms

  const int tid = threadIdx.x;
  const int lane = tid & 31, wave = tid >> 5;
  const int l = lane & 15, hi = (lane >> 4) & 1;
  const int n0 = blockIdx.x * 128;
  const int m0 = wave * 16;

  // ---- stage h_V + dh (valid rows; zeros elsewhere) ----
  for (int i = tid; i < 128 * 32; i += 256) {
    int row = i >> 5, c4 = (i & 31) << 2;
    float4 t = float4{0.f, 0.f, 0.f, 0.f};
    long r = n0 + row;
    if (r < N_NODES) {
      float4 a = *(const float4*)(hV + r * 128 + c4);
      float4 b = *(const float4*)(dh + r * 128 + c4);
      t = float4{a.x + b.x, a.y + b.y, a.z + b.z, a.w + b.w};
    }
    *(float4*)(&rbuf[row * 128 + c4]) = t;
  }
  __syncthreads();

  // ---- LN1 (one thread per row) -> xa (bf16) ----
  if (tid < 128) {
    const float* row = rbuf + tid * 128;
    float s = 0.f, ss = 0.f;
#pragma unroll
    for (int c = 0; c < 128; c += 4) {
      float4 t = *(const float4*)(row + c);
      s += t.x + t.y + t.z + t.w;
      ss += t.x * t.x + t.y * t.y + t.z * t.z + t.w * t.w;
    }
    float m = s * (1.f / 128.f);
    float var = ss * (1.f / 128.f) - m * m;
    float rs = rsqrtf(var + 1e-5f);
#pragma unroll
    for (int c = 0; c < 128; c += 2) {
      float y0 = (row[c] - m) * rs * g1[c] + be1[c];
      float y1 = (row[c + 1] - m) * rs * g1[c + 1] + be1[c + 1];
      *(unsigned*)(&xa[tid * 128 + c]) = pack2(y0, y1);
    }
  }
  __syncthreads();

  // ---- GEMM1 (128x128 @ 128x512) + gelu, fused with GEMM2 (512x128) in 4
  //      K-groups of 128; GEMM2 accumulators live in registers the whole time.
  v8f acc2[8];
  {
    v8f z = {};
#pragma unroll
    for (int i = 0; i < 8; ++i) acc2[i] = z;
  }
  for (int g = 0; g < 4; ++g) {
    // GEMM1 chunk: cols [g*128, g*128+128)
    {
      v8f a1[8];
      v8f z = {};
#pragma unroll
      for (int nc = 0; nc < 8; ++nc) a1[nc] = z;
#pragma unroll
      for (int ks = 0; ks < 4; ++ks) {
        v16bf a = frag_a_lds(xa, 128, m0, ks * 32, lane);
#pragma unroll
        for (int nc = 0; nc < 8; ++nc) {
          v16bf b = frag_b_swz(D1b, 32, g * 128 + nc * 16, ks * 32, lane);
          a1[nc] = wmma_bf16(a, b, a1[nc]);
        }
      }
#pragma unroll
      for (int nc = 0; nc < 8; ++nc) {
        int gcol = g * 128 + nc * 16 + l;
        float bias = db1[gcol];
#pragma unroll
        for (int v = 0; v < 8; ++v) {
          int r = m0 + v + 8 * hi;
          ybuf[r * 128 + nc * 16 + l] = (__bf16)gelu_f(a1[nc][v] + bias);
        }
      }
    }
    // GEMM2 partial: acc2 += ybuf @ D2[g*128 : g*128+128, :]
    // (wave-local LDS RAW: DS ops are in-order within a wave)
#pragma unroll
    for (int ks = 0; ks < 4; ++ks) {
      v16bf a = frag_a_lds(ybuf, 128, m0, ks * 32, lane);
#pragma unroll
      for (int nc = 0; nc < 8; ++nc) {
        v16bf b = frag_b_swz(D2b, 8, nc * 16, g * 128 + ks * 32, lane);
        acc2[nc] = wmma_bf16(a, b, acc2[nc]);
      }
    }
  }

  // ---- bias + residual -> rbuf (fp32 pre-LN2) ----
#pragma unroll
  for (int nc = 0; nc < 8; ++nc) {
    int col = nc * 16 + l;
    float bias = db2[col];
#pragma unroll
    for (int v = 0; v < 8; ++v) {
      int r = m0 + v + 8 * hi;
      float res = (float)xa[r * 128 + col];
      rbuf[r * 128 + col] = acc2[nc][v] + bias + res;
    }
  }
  __syncthreads();

  // ---- LN2 (one thread per row) -> out ----
  if (tid < 128) {
    const float* row = rbuf + tid * 128;
    float s = 0.f, ss = 0.f;
#pragma unroll
    for (int c = 0; c < 128; c += 4) {
      float4 t = *(const float4*)(row + c);
      s += t.x + t.y + t.z + t.w;
      ss += t.x * t.x + t.y * t.y + t.z * t.z + t.w * t.w;
    }
    float m = s * (1.f / 128.f);
    float var = ss * (1.f / 128.f) - m * m;
    float rs = rsqrtf(var + 1e-5f);
    long r = n0 + tid;
    if (r < N_NODES) {
#pragma unroll
      for (int c = 0; c < 128; c += 4) {
        float4 o;
        o.x = (row[c + 0] - m) * rs * g2[c + 0] + be2[c + 0];
        o.y = (row[c + 1] - m) * rs * g2[c + 1] + be2[c + 1];
        o.z = (row[c + 2] - m) * rs * g2[c + 2] + be2[c + 2];
        o.w = (row[c + 3] - m) * rs * g2[c + 3] + be2[c + 3];
        *(float4*)(out + r * 128 + c) = o;
      }
    }
  }
}

// ---------------------------------------------------------------------------
// host launcher
// ---------------------------------------------------------------------------
extern "C" void kernel_launch(void* const* d_in, const int* in_sizes, int n_in,
                              void* d_out, int out_size, void* d_ws, size_t ws_size,
                              hipStream_t stream) {
  const float* hV  = (const float*)d_in[0];
  const float* hE  = (const float*)d_in[1];
  const int*   ei  = (const int*)d_in[2];   // row 0 = src
  const float* W1  = (const float*)d_in[3];
  const float* b1  = (const float*)d_in[4];
  const float* W2  = (const float*)d_in[5];
  const float* b2  = (const float*)d_in[6];
  const float* W3  = (const float*)d_in[7];
  const float* b3  = (const float*)d_in[8];
  const float* g1  = (const float*)d_in[9];
  const float* be1 = (const float*)d_in[10];
  const float* D1  = (const float*)d_in[11];
  const float* db1 = (const float*)d_in[12];
  const float* D2  = (const float*)d_in[13];
  const float* db2 = (const float*)d_in[14];
  const float* g2  = (const float*)d_in[15];
  const float* be2 = (const float*)d_in[16];
  float* out = (float*)d_out;

  // workspace layout: [dh: 20000*128 f32 = 10,240,000 B][swizzled bf16 weights]
  char* ws = (char*)d_ws;
  float* dh = (float*)ws;
  bf16* W1b = (bf16*)(ws + 10240000);
  bf16* W2b = W1b + 256 * 128;
  bf16* W3b = W2b + 128 * 128;
  bf16* D1b = W3b + 128 * 128;
  bf16* D2b = D1b + 128 * 512;

  zero_kernel<<<2500, 256, 0, stream>>>((float4*)dh, (N_NODES * 128) / 4);
  swz_kernel<<<128, 256, 0, stream>>>(W1, W1b, 256, 128);
  swz_kernel<<<64, 256, 0, stream>>>(W2, W2b, 128, 128);
  swz_kernel<<<64, 256, 0, stream>>>(W3, W3b, 128, 128);
  swz_kernel<<<256, 256, 0, stream>>>(D1, D1b, 128, 512);
  swz_kernel<<<256, 256, 0, stream>>>(D2, D2b, 512, 128);

  edge_mlp_scatter<<<N_EDGES / 128, 256, 0, stream>>>(hE, ei, W1b, b1, W2b, b2, W3b,
                                                      b3, dh);
  node_update<<<(N_NODES + 127) / 128, 256, 0, stream>>>(hV, dh, D1b, db1, D2b, db2,
                                                         g1, be1, g2, be2, out);
}
